// HyperBlock_42597485641941
// MI455X (gfx1250) — matrix operand order repaired and checked by reading, not compiled, not run on hardware
//
#include <hip/hip_runtime.h>
#include <cstddef>

typedef __bf16 v16bf __attribute__((ext_vector_type(16)));
typedef float  v8f   __attribute__((ext_vector_type(8)));

union Frag {
  uint4  q[2];
  v16bf  v;
  __bf16 h[16];
};

union FragF {
  float4 q[4];
  float  f[16];
};

// ------------------------------------------------------------------
// fp32 -> bf16 conversion (obs_rep -> O_bf16), 4 elements per thread
// ------------------------------------------------------------------
__global__ void cvt_bf16_kernel(const float* __restrict__ in,
                                __bf16* __restrict__ out, int n4) {
  int i = blockIdx.x * blockDim.x + threadIdx.x;
  if (i >= n4) return;
  const float4 v = reinterpret_cast<const float4*>(in)[i];
  __bf16* o = out + (size_t)i * 4;
  o[0] = (__bf16)v.x; o[1] = (__bf16)v.y; o[2] = (__bf16)v.z; o[3] = (__bf16)v.w;
}

// ------------------------------------------------------------------
// Hypernet first-layer / wide-bias GEMM (N compile-time, multiple of 128):
//   out[1024,N] = act(A[1024,1024](bf16) @ B[1024,N](f32) + bias[N])
// grid: (8, N/128), block 256 (8 waves). Double-buffered B slab in LDS.
// ------------------------------------------------------------------
template <int N>
__global__ __launch_bounds__(256)
void hyper_gemm_kernel(const __bf16* __restrict__ A,
                       const float*  __restrict__ B,
                       const float*  __restrict__ bias,
                       float* __restrict__ out,
                       int do_relu) {
  __shared__ __bf16 BST[2][128 * 40];  // B tile transposed [n][K], pitch 40
  const int t    = threadIdx.x;
  const int lane = t & 31;
  const int wv   = t >> 5;
  const int wm   = wv >> 2;   // 0..1  : M half (64 rows)
  const int wn   = wv & 3;    // 0..3  : N quarter (32 cols)
  const int tl   = lane & 15;
  const int th   = lane >> 4;
  const int mBase = blockIdx.x * 128;
  const int j0    = blockIdx.y * 128;
  const int stN  = t & 127;
  const int stKh = t >> 7;

  const v8f z8 = {0.f,0.f,0.f,0.f,0.f,0.f,0.f,0.f};
  v8f acc[4][2];
#pragma unroll
  for (int i = 0; i < 4; ++i)
#pragma unroll
    for (int j = 0; j < 2; ++j) acc[i][j] = z8;

  for (int kc = 0; kc < 32; ++kc) {
    const int k0 = kc * 32;
    const int p  = kc & 1;
    // stage B chunk [32 x 128] fp32 -> bf16, transposed into LDS
    Frag bt;
    {
      const float* bp = B + (size_t)(k0 + stKh * 16) * N + j0 + stN;
#pragma unroll
      for (int e = 0; e < 16; ++e) bt.h[e] = (__bf16)bp[e * N];  // imm offsets
    }
    *reinterpret_cast<uint4*>(&BST[p][stN * 40 + stKh * 16])     = bt.q[0];
    *reinterpret_cast<uint4*>(&BST[p][stN * 40 + stKh * 16 + 8]) = bt.q[1];
    __syncthreads();

    // A fragments straight from global bf16 (16-bit A layout:
    // e<8 -> K=k0+8*th+e ; e>=8 -> K=k0+8*th+8+e)
    Frag af[4];
#pragma unroll
    for (int ms = 0; ms < 4; ++ms) {
      const int m = mBase + wm * 64 + ms * 16 + tl;
      const __bf16* ap = A + (size_t)m * 1024 + k0 + 8 * th;
      af[ms].q[0] = *reinterpret_cast<const uint4*>(ap);
      af[ms].q[1] = *reinterpret_cast<const uint4*>(ap + 16);
    }
    Frag bfrag[2];
#pragma unroll
    for (int ns = 0; ns < 2; ++ns) {
      const int n = wn * 32 + ns * 16 + tl;
      bfrag[ns].q[0] = *reinterpret_cast<const uint4*>(&BST[p][n * 40 + th * 8]);
      bfrag[ns].q[1] = *reinterpret_cast<const uint4*>(&BST[p][n * 40 + 16 + th * 8]);
    }
#pragma unroll
    for (int ms = 0; ms < 4; ++ms)
#pragma unroll
      for (int ns = 0; ns < 2; ++ns)
        acc[ms][ns] = __builtin_amdgcn_wmma_f32_16x16x32_bf16(
            false, af[ms].v, false, bfrag[ns].v, (short)0, acc[ms][ns], false, false);
  }

#pragma unroll
  for (int ms = 0; ms < 4; ++ms)
#pragma unroll
    for (int ns = 0; ns < 2; ++ns) {
      const int nG = j0 + wn * 32 + ns * 16 + tl;
      const float bv = bias[nG];
#pragma unroll
      for (int vv = 0; vv < 8; ++vv) {
        const int mG = mBase + wm * 64 + ms * 16 + vv + 8 * th;
        float val = acc[ms][ns][vv] + bv;
        if (do_relu) val = fmaxf(val, 0.f);
        out[(size_t)mG * N + nG] = val;
      }
    }
}

// ------------------------------------------------------------------
// Narrow bias heads (pure fp32 dot, trivial FLOPs):
//   b12v[b,c] = O[b,:] @ hb12_w[:,c] + hb12_b[c]   (c<8)
//   b1v[b]    = O[b,:] @ hb1_w[:,0]  + hb1_b[0]    (c==8)
// ------------------------------------------------------------------
__global__ void bias_small_kernel(const float* __restrict__ O,
                                  const float* __restrict__ hb12_w,
                                  const float* __restrict__ hb12_b,
                                  const float* __restrict__ hb1_w,
                                  const float* __restrict__ hb1_b,
                                  float* __restrict__ b12v,
                                  float* __restrict__ b1v) {
  int idx = blockIdx.x * blockDim.x + threadIdx.x;  // 1024*16 (c in 0..8 used)
  if (idx >= 1024 * 16) return;
  const int b = idx >> 4, c = idx & 15;
  if (c > 8) return;
  const float* o = O + (size_t)b * 1024;
  float s;
  if (c < 8) {
    s = hb12_b[c];
    for (int kk = 0; kk < 1024; ++kk) s += o[kk] * hb12_w[kk * 8 + c];
    b12v[b * 8 + c] = s;
  } else {
    s = hb1_b[0];
    for (int kk = 0; kk < 1024; ++kk) s += o[kk] * hb1_w[kk];
    b1v[b] = s;
  }
}

// ------------------------------------------------------------------
// Tiny heads: w1 = RH1@hw1_w2 + hw1_b2 ; w12 = RH12@hw12_w2 + hw12_b2
// ------------------------------------------------------------------
__global__ void heads_kernel(const float* __restrict__ RH1,
                             const float* __restrict__ RH12,
                             const float* __restrict__ w1w2,
                             const float* __restrict__ w1b2,
                             const float* __restrict__ w12w2,
                             const float* __restrict__ w12b2,
                             float* __restrict__ w1o,
                             float* __restrict__ w12o) {
  int idx = blockIdx.x * blockDim.x + threadIdx.x;  // 1024*16
  if (idx >= 1024 * 16) return;
  const int b = idx >> 4, c = idx & 15, j = c & 7;
  const float* rh = (c < 8) ? (RH1 + (size_t)b * 128) : (RH12 + (size_t)b * 128);
  const float* W  = (c < 8) ? w1w2 : w12w2;
  float s = (c < 8) ? w1b2[j] : w12b2[j];
  for (int kk = 0; kk < 128; ++kk) s += rh[kk] * W[kk * 8 + j];
  if (c < 8) w1o[b * 8 + j] = s; else w12o[b * 8 + j] = s;
}

// ------------------------------------------------------------------
// Stages 1+2 (per-sample tiny matmuls, pure elementwise cost):
//   hid1[b,i] = relu( sum_j x[b, i*8+j] * w1[b,j] + b1[b] )
//   H2[b, i*8+j] = relu( hid1[b,i]*w12[b,j] + b12[b,j] )
// ------------------------------------------------------------------
__global__ void stage12_kernel(const float* __restrict__ x,
                               const float* __restrict__ w1,
                               const float* __restrict__ b1v,
                               const float* __restrict__ w12,
                               const float* __restrict__ b12v,
                               float* __restrict__ H2) {
  int idx = blockIdx.x * blockDim.x + threadIdx.x;  // 1024*128
  if (idx >= 1024 * 128) return;
  const int b = idx >> 7, i = idx & 127;
  const float* xr  = x + (size_t)b * 1024 + i * 8;
  const float* w1r = w1 + b * 8;
  float s = b1v[b];
#pragma unroll
  for (int j = 0; j < 8; ++j) s += xr[j] * w1r[j];
  const float h = fmaxf(s, 0.f);
  const float* w12r = w12 + b * 8;
  const float* b12r = b12v + b * 8;
  float* o = H2 + (size_t)b * 1024 + i * 8;
#pragma unroll
  for (int j = 0; j < 8; ++j) o[j] = fmaxf(h * w12r[j] + b12r[j], 0.f);
}

// ------------------------------------------------------------------
// Fused implicit-A GEMM for stages 3 / 4 (ID, JD compile-time):
//   out[(b,a), j] = relu( sum_{k<=128, i<ID} rhk * H[(b,a),i] * Wrow[k,i][j]
//                         + bias2[b, j] )  (+ res)   rhk = RH[b,k] (k<128), 1 (k==128)
//   Wrow = W + (k*ID+i)*JD (k<128) | BW + i*JD (bias fold, k==128)
// A tile generated on the fly: A[(b,a), (k,i)] = RH[b,k] * H[(b,a), i]
// grid: (64, JD/128), block 256. Double-buffered B slab; H slab in fp32.
// ------------------------------------------------------------------
template <int ID, int JD, bool RES>
__global__ __launch_bounds__(256)
void hyper_fused_kernel(const float* __restrict__ H,     // [8192, ID]
                        const float* __restrict__ RH,    // [1024, 128]
                        const float* __restrict__ W,     // [128 * ID * JD]
                        const float* __restrict__ BW,    // [ID * JD]
                        const float* __restrict__ bias2, // [1024, JD]
                        const float* __restrict__ res,   // [8192, JD] (if RES)
                        float* __restrict__ out) {       // [8192, JD]
  __shared__ float  rhS[16 * 129];     // rh per (local sample, k)
  __shared__ float  HS[128 * 36];      // H slab fp32 [m][i_local], pitch 36
  __shared__ __bf16 BST[2][128 * 40];  // B slab transposed [n][K_local], pitch 40
  const int t    = threadIdx.x;
  const int lane = t & 31;
  const int wv   = t >> 5;
  const int wm   = wv >> 2;
  const int wn   = wv & 3;
  const int tl   = lane & 15;
  const int th   = lane >> 4;
  const int mBase = blockIdx.x * 128;
  const int bBase = blockIdx.x * 16;
  const int j0    = blockIdx.y * 128;
  const int stN  = t & 127;
  const int stKh = t >> 7;
  const int hM   = t >> 1;
  const int hI   = (t & 1) * 16;

  for (int idx = t; idx < 2048; idx += 256)
    rhS[(idx >> 7) * 129 + (idx & 127)] =
        RH[(size_t)(bBase + (idx >> 7)) * 128 + (idx & 127)];

  const v8f z8 = {0.f,0.f,0.f,0.f,0.f,0.f,0.f,0.f};
  v8f acc[4][2];
#pragma unroll
  for (int i = 0; i < 4; ++i)
#pragma unroll
    for (int j = 0; j < 2; ++j) acc[i][j] = z8;

  int cc = 0;  // chunk counter (LDS B-buffer parity)
  for (int i0 = 0; i0 < ID; i0 += 32) {
    // stage H slab [128 m][32 i] fp32 (pure copies, no conversion)
    FragF ht;
    {
      const float* hp = H + (size_t)(mBase + hM) * ID + i0 + hI;
#pragma unroll
      for (int q = 0; q < 4; ++q)
        ht.q[q] = *reinterpret_cast<const float4*>(hp + 4 * q);
    }
    __syncthreads();  // previous chunk's HS readers done
#pragma unroll
    for (int q = 0; q < 4; ++q)
      *reinterpret_cast<float4*>(&HS[hM * 36 + hI + 4 * q]) = ht.q[q];

    for (int k = 0; k <= 128; ++k, ++cc) {
      const int p = cc & 1;
      const float* base = (k < 128) ? (W + ((size_t)k * ID + i0) * JD + j0)
                                    : (BW + (size_t)i0 * JD + j0);
      const float* bp = base + stN;
      Frag bt;
#pragma unroll
      for (int e = 0; e < 16; ++e)
        bt.h[e] = (__bf16)bp[(stKh * 16 + e) * JD];  // imm offsets (JD constexpr)
      if (k < 127)
        __builtin_prefetch(base + (size_t)ID * JD + stN, 0, 1);  // next k's rows
      *reinterpret_cast<uint4*>(&BST[p][stN * 40 + stKh * 16])     = bt.q[0];
      *reinterpret_cast<uint4*>(&BST[p][stN * 40 + stKh * 16 + 8]) = bt.q[1];
      __syncthreads();

      // generate A fragments: bf16( rh[b,k] * H[m, i0+K_local] )
      Frag af[4];
#pragma unroll
      for (int ms = 0; ms < 4; ++ms) {
        const int mL = wm * 64 + ms * 16 + tl;
        const float rh = (k < 128) ? rhS[(mL >> 3) * 129 + k] : 1.0f;
        FragF hfr;
        hfr.q[0] = *reinterpret_cast<const float4*>(&HS[mL * 36 + 8 * th]);
        hfr.q[1] = *reinterpret_cast<const float4*>(&HS[mL * 36 + 8 * th + 4]);
        hfr.q[2] = *reinterpret_cast<const float4*>(&HS[mL * 36 + 16 + 8 * th]);
        hfr.q[3] = *reinterpret_cast<const float4*>(&HS[mL * 36 + 16 + 8 * th + 4]);
#pragma unroll
        for (int e = 0; e < 16; ++e)
          af[ms].h[e] = (__bf16)(rh * hfr.f[e]);
      }
      Frag bfrag[2];
#pragma unroll
      for (int ns = 0; ns < 2; ++ns) {
        const int n = wn * 32 + ns * 16 + tl;
        bfrag[ns].q[0] = *reinterpret_cast<const uint4*>(&BST[p][n * 40 + th * 8]);
        bfrag[ns].q[1] = *reinterpret_cast<const uint4*>(&BST[p][n * 40 + 16 + th * 8]);
      }
#pragma unroll
      for (int ms = 0; ms < 4; ++ms)
#pragma unroll
        for (int ns = 0; ns < 2; ++ns)
          acc[ms][ns] = __builtin_amdgcn_wmma_f32_16x16x32_bf16(
              false, af[ms].v, false, bfrag[ns].v, (short)0, acc[ms][ns], false, false);
    }
  }

  // epilogue: + per-sample bias, relu, optional residual add (x)
#pragma unroll
  for (int ms = 0; ms < 4; ++ms)
#pragma unroll
    for (int ns = 0; ns < 2; ++ns) {
      const int nG = j0 + wn * 32 + ns * 16 + tl;
#pragma unroll
      for (int vv = 0; vv < 8; ++vv) {
        const int mG = mBase + wm * 64 + ms * 16 + vv + 8 * th;
        float val = acc[ms][ns][vv] + bias2[(size_t)(mG >> 3) * JD + nG];
        val = fmaxf(val, 0.f);
        if (RES) val += res[(size_t)mG * JD + nG];
        out[(size_t)mG * JD + nG] = val;
      }
    }
}

// ------------------------------------------------------------------
extern "C" void kernel_launch(void* const* d_in, const int* in_sizes, int n_in,
                              void* d_out, int out_size, void* d_ws, size_t ws_size,
                              hipStream_t stream) {
  (void)in_sizes; (void)n_in; (void)out_size; (void)ws_size;
  const float* x       = (const float*)d_in[0];
  const float* obs     = (const float*)d_in[1];
  const float* hw1_w1  = (const float*)d_in[2];
  const float* hw1_b1  = (const float*)d_in[3];
  const float* hw1_w2  = (const float*)d_in[4];
  const float* hw1_b2  = (const float*)d_in[5];
  const float* hw12_w1 = (const float*)d_in[6];
  const float* hw12_b1 = (const float*)d_in[7];
  const float* hw12_w2 = (const float*)d_in[8];
  const float* hw12_b2 = (const float*)d_in[9];
  const float* hw2_w1  = (const float*)d_in[10];
  const float* hw2_b1  = (const float*)d_in[11];
  const float* hw2_w2  = (const float*)d_in[12];
  const float* hw2_b2  = (const float*)d_in[13];
  const float* hw22_w1 = (const float*)d_in[14];
  const float* hw22_b1 = (const float*)d_in[15];
  const float* hw22_w2 = (const float*)d_in[16];
  const float* hw22_b2 = (const float*)d_in[17];
  const float* hb1_w   = (const float*)d_in[18];
  const float* hb1_b   = (const float*)d_in[19];
  const float* hb12_w  = (const float*)d_in[20];
  const float* hb12_b  = (const float*)d_in[21];
  const float* hb2_w   = (const float*)d_in[22];
  const float* hb2_b   = (const float*)d_in[23];
  const float* hb22_w  = (const float*)d_in[24];
  const float* hb22_b  = (const float*)d_in[25];

  char* wp = (char*)d_ws;
  auto alloc = [&](size_t bytes) {
    char* p = wp; wp += (bytes + 255) & ~(size_t)255; return p;
  };
  __bf16* Obf = (__bf16*)alloc((size_t)1024 * 1024 * 2);
  float* RH1  = (float*)alloc((size_t)1024 * 128 * 4);
  float* RH12 = (float*)alloc((size_t)1024 * 128 * 4);
  float* RH2  = (float*)alloc((size_t)1024 * 128 * 4);
  float* RH22 = (float*)alloc((size_t)1024 * 128 * 4);
  float* b2v  = (float*)alloc((size_t)1024 * 512 * 4);
  float* b22v = (float*)alloc((size_t)1024 * 128 * 4);
  float* b12v = (float*)alloc((size_t)1024 * 8 * 4);
  float* b1v  = (float*)alloc((size_t)1024 * 4);
  float* w1v  = (float*)alloc((size_t)1024 * 8 * 4);
  float* w12v = (float*)alloc((size_t)1024 * 8 * 4);
  float* H2   = (float*)alloc((size_t)8192 * 128 * 4);
  float* H3   = (float*)alloc((size_t)8192 * 512 * 4);

  cvt_bf16_kernel<<<1024, 256, 0, stream>>>(obs, Obf, 1024 * 1024 / 4);

  hyper_gemm_kernel<128><<<dim3(8, 1), 256, 0, stream>>>(Obf, hw1_w1,  hw1_b1,  RH1,  1);
  hyper_gemm_kernel<128><<<dim3(8, 1), 256, 0, stream>>>(Obf, hw12_w1, hw12_b1, RH12, 1);
  hyper_gemm_kernel<128><<<dim3(8, 1), 256, 0, stream>>>(Obf, hw2_w1,  hw2_b1,  RH2,  1);
  hyper_gemm_kernel<128><<<dim3(8, 1), 256, 0, stream>>>(Obf, hw22_w1, hw22_b1, RH22, 1);
  hyper_gemm_kernel<512><<<dim3(8, 4), 256, 0, stream>>>(Obf, hb2_w,   hb2_b,   b2v,  0);
  hyper_gemm_kernel<128><<<dim3(8, 1), 256, 0, stream>>>(Obf, hb22_w,  hb22_b,  b22v, 0);

  bias_small_kernel<<<64, 256, 0, stream>>>(obs, hb12_w, hb12_b, hb1_w, hb1_b, b12v, b1v);
  heads_kernel<<<64, 256, 0, stream>>>(RH1, RH12, hw1_w2, hw1_b2, hw12_w2, hw12_b2, w1v, w12v);
  stage12_kernel<<<512, 256, 0, stream>>>(x, w1v, b1v, w12v, b12v, H2);

  // stage 3: H3[8192,512] = relu(G @ W2v + biases)
  hyper_fused_kernel<128, 512, false><<<dim3(64, 4), 256, 0, stream>>>(
      H2, RH2, hw2_w2, hw2_b2, b2v, nullptr, H3);
  // stage 4: out = x + relu(G4 @ W22v + biases)
  hyper_fused_kernel<512, 128, true><<<dim3(64, 1), 256, 0, stream>>>(
      H3, RH22, hw22_w2, hw22_b2, b22v, x, (float*)d_out);
}
